// MoE_89988154785975
// MI455X (gfx1250) — compile-verified, matrix operand order checked by reading, not run on hardware
//
#include <hip/hip_runtime.h>
#include <math.h>

// ---------------------------------------------------------------------------
// Problem constants (from reference): B=2,T=2048,D=1024,F=4096,E=8,K=2
// ---------------------------------------------------------------------------
#define Bc 2
#define Tc 2048
#define Dc 1024
#define Fc 4096
#define Ec 8
#define Kc 2
#define BT (Bc * Tc)          // 4096 tokens
#define TOT_SLOTS (BT * Kc)   // 8192 token-expert slots

// GEMM tiling: block = 256 threads = 8 waves (2 Mwaves x 4 Nwaves)
// wave tile 64x64 -> 4x4 fragments of v_wmma_f32_16x16x32_bf16
#define BLK_M 128
#define BLK_N 256
#define LDSTR 40   // padded LDS row stride in halves (80B, multiple of 16B)

typedef __bf16 v16bf __attribute__((ext_vector_type(16)));
typedef float  v8f   __attribute__((ext_vector_type(8)));

union FragBF {
    v16bf v;
    uint4 q[2];
};

__device__ __forceinline__ unsigned short bf16_rne(float f) {
    unsigned int u = __float_as_uint(f);
    unsigned int lsb = (u >> 16) & 1u;
    u += 0x7FFFu + lsb;
    return (unsigned short)(u >> 16);
}

// ---------------------------------------------------------------------------
// gfx1250 async global->LDS copy (ASYNCcnt path); fallback = VGPR round trip
// Builtin parameter types (from hipcc diagnostic): v4i addrspace(1)* src,
// lds dst, imm offset, imm cpol.
// ---------------------------------------------------------------------------
#if __has_builtin(__builtin_amdgcn_global_load_async_to_lds_b128) && \
    __has_builtin(__builtin_amdgcn_s_wait_asynccnt)
#define MOE_ASYNC_LDS 1
#endif

typedef int moe_v4i __attribute__((vector_size(16)));
typedef __attribute__((address_space(1))) moe_v4i* moe_gptr;
typedef __attribute__((address_space(3))) moe_v4i* moe_lptr;

__device__ __forceinline__ void lds_copy16(void* lds_dst, const void* gsrc) {
#ifdef MOE_ASYNC_LDS
    __builtin_amdgcn_global_load_async_to_lds_b128(
        (moe_gptr)gsrc, (moe_lptr)lds_dst, 0, 0);
#else
    *(uint4*)lds_dst = *(const uint4*)gsrc;
#endif
}

// Wait until at most `n` async copies remain in flight for this wave.
// ASYNCcnt decrements in order, so waiting to <=6 after issuing 6 new copies
// guarantees the previous buffer's 6 copies have landed.
template <int N>
__device__ __forceinline__ void lds_copy_wait() {
#ifdef MOE_ASYNC_LDS
    __builtin_amdgcn_s_wait_asynccnt(N);
#endif
}

// ---------------------------------------------------------------------------
// 0) zero the per-expert counters
// ---------------------------------------------------------------------------
__global__ void moe_zero_counts(int* counts, int* cursor) {
    int t = threadIdx.x;
    if (t < Ec) { counts[t] = 0; cursor[t] = 0; }
}

// ---------------------------------------------------------------------------
// 1) convert x (f32) -> xh (bf16), 4 elements / thread
// ---------------------------------------------------------------------------
__global__ void moe_conv_x(const float* __restrict__ x, unsigned short* __restrict__ xh) {
    int i = (blockIdx.x * blockDim.x + threadIdx.x) * 4;
    if (i >= BT * Dc) return;
    float4 v = *(const float4*)(x + i);
    unsigned int lo = (unsigned int)bf16_rne(v.x) | ((unsigned int)bf16_rne(v.y) << 16);
    unsigned int hi = (unsigned int)bf16_rne(v.z) | ((unsigned int)bf16_rne(v.w) << 16);
    uint2 o; o.x = lo; o.y = hi;
    *(uint2*)(xh + i) = o;
}

// ---------------------------------------------------------------------------
// 2) batched transpose + f32->bf16: dst[e][c][r] = bf16(src[e][r][c])
//    (gives K-contiguous B operands for WMMA). Block 32x8, tile 32x32.
// ---------------------------------------------------------------------------
__global__ void moe_transpose_bf16(const float* __restrict__ src,
                                   unsigned short* __restrict__ dst,
                                   int R, int C) {
    __shared__ float tile[32][33];
    const size_t batch = (size_t)blockIdx.z * (size_t)R * (size_t)C;
    src += batch; dst += batch;
    const int rb = blockIdx.y * 32, cb = blockIdx.x * 32;
#pragma unroll
    for (int i = 0; i < 4; ++i) {
        int r = rb + threadIdx.y + i * 8;
        tile[threadIdx.y + i * 8][threadIdx.x] = src[(size_t)r * C + cb + threadIdx.x];
    }
    __syncthreads();
#pragma unroll
    for (int i = 0; i < 4; ++i) {
        int r = cb + threadIdx.y + i * 8;   // row of dst = original column
        int c = rb + threadIdx.x;           // col of dst = original row
        dst[(size_t)r * R + c] = bf16_rne(tile[threadIdx.x][threadIdx.y + i * 8]);
    }
}

// ---------------------------------------------------------------------------
// 3) gating: logits = x·Wg + bg ; softmax ; top-2 ; count tokens per expert
// ---------------------------------------------------------------------------
__global__ void moe_gating(const float* __restrict__ x,
                           const float* __restrict__ Wg,
                           const float* __restrict__ bg,
                           int* __restrict__ counts,
                           int* __restrict__ eks,
                           float* __restrict__ wks) {
    int t = blockIdx.x * blockDim.x + threadIdx.x;
    if (t >= BT) return;
    float acc[Ec];
#pragma unroll
    for (int e = 0; e < Ec; ++e) acc[e] = bg[e];
    const float* xr = x + (size_t)t * Dc;
    for (int d = 0; d < Dc; d += 4) {
        float4 xv = *(const float4*)(xr + d);
#pragma unroll
        for (int e = 0; e < Ec; ++e) {
            acc[e] += xv.x * Wg[(d + 0) * Ec + e];
            acc[e] += xv.y * Wg[(d + 1) * Ec + e];
            acc[e] += xv.z * Wg[(d + 2) * Ec + e];
            acc[e] += xv.w * Wg[(d + 3) * Ec + e];
        }
    }
    // softmax
    float m = acc[0];
#pragma unroll
    for (int e = 1; e < Ec; ++e) m = fmaxf(m, acc[e]);
    float s = 0.f, p[Ec];
#pragma unroll
    for (int e = 0; e < Ec; ++e) { p[e] = __expf(acc[e] - m); s += p[e]; }
    float inv = 1.0f / s;
#pragma unroll
    for (int e = 0; e < Ec; ++e) p[e] *= inv;
    // top-2 (first occurrence wins, matching lax.top_k)
    int i0 = 0;
#pragma unroll
    for (int e = 1; e < Ec; ++e) if (p[e] > p[i0]) i0 = e;
    int i1 = (i0 == 0) ? 1 : 0;
#pragma unroll
    for (int e = 0; e < Ec; ++e) if (e != i0 && p[e] > p[i1]) i1 = e;

    eks[t * 2 + 0] = i0; wks[t * 2 + 0] = p[i0];
    eks[t * 2 + 1] = i1; wks[t * 2 + 1] = p[i1];
    atomicAdd(&counts[i0], 1);
    atomicAdd(&counts[i1], 1);
}

// ---------------------------------------------------------------------------
// 4) prefix-sum offsets (E=8, single thread)
// ---------------------------------------------------------------------------
__global__ void moe_offsets(const int* __restrict__ counts, int* __restrict__ offsets) {
    if (threadIdx.x == 0) {
        int acc = 0;
        for (int e = 0; e < Ec; ++e) { offsets[e] = acc; acc += counts[e]; }
    }
}

// ---------------------------------------------------------------------------
// 5) scatter tokens into compact per-expert buckets
// ---------------------------------------------------------------------------
__global__ void moe_scatter(const int* __restrict__ eks,
                            const int* __restrict__ offsets,
                            int* __restrict__ cursor,
                            int* __restrict__ bucket,
                            int* __restrict__ slotOf) {
    int t = blockIdx.x * blockDim.x + threadIdx.x;
    if (t >= BT) return;
#pragma unroll
    for (int k = 0; k < Kc; ++k) {
        int e = eks[t * 2 + k];
        int cur = atomicAdd(&cursor[e], 1);
        bucket[e * BT + cur] = t;
        slotOf[t * 2 + k] = offsets[e] + cur;
    }
}

// ---------------------------------------------------------------------------
// WMMA GEMM 1:  H[slot, F] = gelu( xh[token]·W1t[e]^T + b1[e] )   (bf16 out)
//   tile BLK_M x BLK_N, K-step 32, double-buffered async global->LDS pipeline
// ---------------------------------------------------------------------------
__global__ __launch_bounds__(256) void moe_ffn_gemm1(
        const unsigned short* __restrict__ xh,
        const unsigned short* __restrict__ W1t,   // [E][F][D] bf16
        const float* __restrict__ b1,             // [E][F]
        const int* __restrict__ counts,
        const int* __restrict__ offsets,
        const int* __restrict__ bucket,
        unsigned short* __restrict__ H) {         // [TOT_SLOTS][F] bf16
    const int e  = blockIdx.z;
    const int mt = blockIdx.y;
    const int nt = blockIdx.x;
    const int cnt = counts[e];
    if (mt * BLK_M >= cnt) return;
    const int off = offsets[e];

    __shared__ __align__(16) unsigned short As[2][BLK_M * LDSTR];
    __shared__ __align__(16) unsigned short Bs[2][BLK_N * LDSTR];
    __shared__ int tok[BLK_M];

    const int tid = threadIdx.x;
    if (tid < BLK_M) {
        int row = mt * BLK_M + tid;
        tok[tid] = bucket[e * BT + ((row < cnt) ? row : 0)];
    }
    __syncthreads();

    const int lane = tid & 31;
    const int wv = tid >> 5;
    const int wm = wv >> 2;          // 0..1 -> M offset wm*64
    const int wn = wv & 3;           // 0..3 -> N offset wn*64
    const int c0 = (lane >> 4) * 8;  // K chunk select per ISA bf16 A/B layout
    const int l15 = lane & 15;

    const unsigned short* Wbase = W1t + (size_t)e * Fc * Dc + (size_t)(nt * BLK_N) * Dc;

    // 6 async b128 copies per thread per stage (2 for A, 4 for B)
    auto stage = [&](int buf, int k0) {
#pragma unroll
        for (int i = 0; i < 2; ++i) {
            int cid = tid + i * 256;
            int r = cid >> 2, c = cid & 3;
            lds_copy16(&As[buf][r * LDSTR + c * 8], xh + (size_t)tok[r] * Dc + k0 + c * 8);
        }
#pragma unroll
        for (int i = 0; i < 4; ++i) {
            int cid = tid + i * 256;
            int r = cid >> 2, c = cid & 3;
            lds_copy16(&Bs[buf][r * LDSTR + c * 8], Wbase + (size_t)r * Dc + k0 + c * 8);
        }
    };

    v8f acc[4][4];
#pragma unroll
    for (int i = 0; i < 4; ++i)
#pragma unroll
        for (int j = 0; j < 4; ++j) acc[i][j] = (v8f)0.0f;

    stage(0, 0);   // prologue fill

    for (int k0 = 0; k0 < Dc; k0 += 32) {
        const int cur = (k0 >> 5) & 1;
        if (k0 + 32 < Dc) {
            stage(cur ^ 1, k0 + 32);   // keep next tile's DMA in flight
            lds_copy_wait<6>();        // current tile's 6 copies have landed
        } else {
            lds_copy_wait<0>();
        }
        __syncthreads();

        FragBF a[4], b[4];
#pragma unroll
        for (int mf = 0; mf < 4; ++mf) {
            int r = wm * 64 + mf * 16 + l15;
            a[mf].q[0] = *(const uint4*)&As[cur][r * LDSTR + c0];
            a[mf].q[1] = *(const uint4*)&As[cur][r * LDSTR + c0 + 16];
        }
#pragma unroll
        for (int nf = 0; nf < 4; ++nf) {
            int n = wn * 64 + nf * 16 + l15;
            b[nf].q[0] = *(const uint4*)&Bs[cur][n * LDSTR + c0];
            b[nf].q[1] = *(const uint4*)&Bs[cur][n * LDSTR + c0 + 16];
        }
#pragma unroll
        for (int mf = 0; mf < 4; ++mf)
#pragma unroll
            for (int nf = 0; nf < 4; ++nf)
                acc[mf][nf] = __builtin_amdgcn_wmma_f32_16x16x32_bf16(
                    false, a[mf].v, false, b[nf].v, (short)0, acc[mf][nf], false, false);
        __syncthreads();   // done reading cur buffer; it may be overwritten next iter
    }

    // epilogue: +b1, exact gelu, bf16 store to H
    const int colbase = nt * BLK_N + wn * 64;
#pragma unroll
    for (int mf = 0; mf < 4; ++mf) {
#pragma unroll
        for (int nf = 0; nf < 4; ++nf) {
            int fcol = colbase + nf * 16 + l15;
            float bias = b1[e * Fc + fcol];
            int rl0 = wm * 64 + mf * 16 + ((lane >> 4) * 8);
#pragma unroll
            for (int r = 0; r < 8; ++r) {
                int mrow = mt * BLK_M + rl0 + r;
                if (mrow < cnt) {
                    float v = acc[mf][nf][r] + bias;
                    v = 0.5f * v * (1.0f + erff(v * 0.70710678118f));
                    H[(size_t)(off + mrow) * Fc + fcol] = bf16_rne(v);
                }
            }
        }
    }
}

// ---------------------------------------------------------------------------
// WMMA GEMM 2:  Y[slot, D] = H[slot]·W2t[e]^T      (f32 out, no bias yet)
// ---------------------------------------------------------------------------
__global__ __launch_bounds__(256) void moe_ffn_gemm2(
        const unsigned short* __restrict__ H,     // [TOT_SLOTS][F] bf16
        const unsigned short* __restrict__ W2t,   // [E][D][F] bf16
        const int* __restrict__ counts,
        const int* __restrict__ offsets,
        float* __restrict__ Y) {                  // [TOT_SLOTS][D] f32
    const int e  = blockIdx.z;
    const int mt = blockIdx.y;
    const int nt = blockIdx.x;
    const int cnt = counts[e];
    if (mt * BLK_M >= cnt) return;
    const int off = offsets[e];

    __shared__ __align__(16) unsigned short As[2][BLK_M * LDSTR];
    __shared__ __align__(16) unsigned short Bs[2][BLK_N * LDSTR];

    const int tid = threadIdx.x;
    const int lane = tid & 31;
    const int wv = tid >> 5;
    const int wm = wv >> 2;
    const int wn = wv & 3;
    const int c0 = (lane >> 4) * 8;
    const int l15 = lane & 15;

    const unsigned short* Wbase = W2t + (size_t)e * Dc * Fc + (size_t)(nt * BLK_N) * Fc;

    auto stage = [&](int buf, int k0) {
#pragma unroll
        for (int i = 0; i < 2; ++i) {
            int cid = tid + i * 256;
            int r = cid >> 2, c = cid & 3;
            int rr = mt * BLK_M + r; rr = (rr < cnt) ? rr : (cnt - 1);
            lds_copy16(&As[buf][r * LDSTR + c * 8], H + (size_t)(off + rr) * Fc + k0 + c * 8);
        }
#pragma unroll
        for (int i = 0; i < 4; ++i) {
            int cid = tid + i * 256;
            int r = cid >> 2, c = cid & 3;
            lds_copy16(&Bs[buf][r * LDSTR + c * 8], Wbase + (size_t)r * Fc + k0 + c * 8);
        }
    };

    v8f acc[4][4];
#pragma unroll
    for (int i = 0; i < 4; ++i)
#pragma unroll
        for (int j = 0; j < 4; ++j) acc[i][j] = (v8f)0.0f;

    stage(0, 0);

    for (int k0 = 0; k0 < Fc; k0 += 32) {
        const int cur = (k0 >> 5) & 1;
        if (k0 + 32 < Fc) {
            stage(cur ^ 1, k0 + 32);
            lds_copy_wait<6>();
        } else {
            lds_copy_wait<0>();
        }
        __syncthreads();

        FragBF a[4], b[4];
#pragma unroll
        for (int mf = 0; mf < 4; ++mf) {
            int r = wm * 64 + mf * 16 + l15;
            a[mf].q[0] = *(const uint4*)&As[cur][r * LDSTR + c0];
            a[mf].q[1] = *(const uint4*)&As[cur][r * LDSTR + c0 + 16];
        }
#pragma unroll
        for (int nf = 0; nf < 4; ++nf) {
            int n = wn * 64 + nf * 16 + l15;
            b[nf].q[0] = *(const uint4*)&Bs[cur][n * LDSTR + c0];
            b[nf].q[1] = *(const uint4*)&Bs[cur][n * LDSTR + c0 + 16];
        }
#pragma unroll
        for (int mf = 0; mf < 4; ++mf)
#pragma unroll
            for (int nf = 0; nf < 4; ++nf)
                acc[mf][nf] = __builtin_amdgcn_wmma_f32_16x16x32_bf16(
                    false, a[mf].v, false, b[nf].v, (short)0, acc[mf][nf], false, false);
        __syncthreads();
    }

    const int colbase = nt * BLK_N + wn * 64;
#pragma unroll
    for (int mf = 0; mf < 4; ++mf) {
#pragma unroll
        for (int nf = 0; nf < 4; ++nf) {
            int dcol = colbase + nf * 16 + l15;
            int rl0 = wm * 64 + mf * 16 + ((lane >> 4) * 8);
#pragma unroll
            for (int r = 0; r < 8; ++r) {
                int mrow = mt * BLK_M + rl0 + r;
                if (mrow < cnt)
                    Y[(size_t)(off + mrow) * Dc + dcol] = acc[mf][nf][r];
            }
        }
    }
}

// ---------------------------------------------------------------------------
// 6) combine: out[t] = w0*(Y[s0]+b2[e0]) + w1*(Y[s1]+b2[e1])  (deterministic)
// ---------------------------------------------------------------------------
__global__ void moe_combine(const float* __restrict__ Y,
                            const float* __restrict__ b2,
                            const int* __restrict__ eks,
                            const float* __restrict__ wks,
                            const int* __restrict__ slotOf,
                            float* __restrict__ out) {
    int gid = blockIdx.x * blockDim.x + threadIdx.x;
    const int chunks = Dc / 4;
    if (gid >= BT * chunks) return;
    int t = gid / chunks;
    int d = (gid % chunks) * 4;
    int s0 = slotOf[t * 2 + 0], s1 = slotOf[t * 2 + 1];
    int e0 = eks[t * 2 + 0],    e1 = eks[t * 2 + 1];
    float w0 = wks[t * 2 + 0],  w1 = wks[t * 2 + 1];
    float4 y0 = *(const float4*)(Y + (size_t)s0 * Dc + d);
    float4 y1 = *(const float4*)(Y + (size_t)s1 * Dc + d);
    float4 c0 = *(const float4*)(b2 + (size_t)e0 * Dc + d);
    float4 c1 = *(const float4*)(b2 + (size_t)e1 * Dc + d);
    float4 o;
    o.x = w0 * (y0.x + c0.x) + w1 * (y1.x + c1.x);
    o.y = w0 * (y0.y + c0.y) + w1 * (y1.y + c1.y);
    o.z = w0 * (y0.z + c0.z) + w1 * (y1.z + c1.z);
    o.w = w0 * (y0.w + c0.w) + w1 * (y1.w + c1.w);
    *(float4*)(out + (size_t)t * Dc + d) = o;
}

// ---------------------------------------------------------------------------
// launcher
// ---------------------------------------------------------------------------
extern "C" void kernel_launch(void* const* d_in, const int* in_sizes, int n_in,
                              void* d_out, int out_size, void* d_ws, size_t ws_size,
                              hipStream_t stream) {
    (void)in_sizes; (void)n_in; (void)out_size; (void)ws_size;
    const float* x  = (const float*)d_in[0];
    const float* Wg = (const float*)d_in[1];
    const float* bg = (const float*)d_in[2];
    const float* W1 = (const float*)d_in[3];
    const float* b1 = (const float*)d_in[4];
    const float* W2 = (const float*)d_in[5];
    const float* b2 = (const float*)d_in[6];
    float* out = (float*)d_out;

    // workspace carve-up (256B aligned)
    char* ws = (char*)d_ws;
    size_t o = 0;
    auto carve = [&](size_t bytes) { void* p = ws + o; o = (o + bytes + 255) & ~(size_t)255; return p; };
    unsigned short* xh   = (unsigned short*)carve((size_t)BT * Dc * 2);          // 8 MB
    unsigned short* W1t  = (unsigned short*)carve((size_t)Ec * Dc * Fc * 2);     // 64 MB
    unsigned short* W2t  = (unsigned short*)carve((size_t)Ec * Fc * Dc * 2);     // 64 MB
    unsigned short* H    = (unsigned short*)carve((size_t)TOT_SLOTS * Fc * 2);   // 64 MB
    float*          Yb   = (float*)carve((size_t)TOT_SLOTS * Dc * 4);            // 32 MB
    int*   counts  = (int*)carve(Ec * 4);
    int*   cursor  = (int*)carve(Ec * 4);
    int*   offsets = (int*)carve(Ec * 4);
    int*   bucket  = (int*)carve((size_t)Ec * BT * 4);
    int*   eks     = (int*)carve((size_t)BT * 2 * 4);
    float* wks     = (float*)carve((size_t)BT * 2 * 4);
    int*   slotOf  = (int*)carve((size_t)BT * 2 * 4);

    hipLaunchKernelGGL(moe_zero_counts, dim3(1), dim3(32), 0, stream, counts, cursor);
    hipLaunchKernelGGL(moe_conv_x, dim3((BT * Dc / 4 + 255) / 256), dim3(256), 0, stream, x, xh);
    // W1: [E][D][F] -> W1t [E][F][D]
    hipLaunchKernelGGL(moe_transpose_bf16, dim3(Fc / 32, Dc / 32, Ec), dim3(32, 8), 0, stream,
                       W1, W1t, Dc, Fc);
    // W2: [E][F][D] -> W2t [E][D][F]
    hipLaunchKernelGGL(moe_transpose_bf16, dim3(Dc / 32, Fc / 32, Ec), dim3(32, 8), 0, stream,
                       W2, W2t, Fc, Dc);
    hipLaunchKernelGGL(moe_gating, dim3((BT + 255) / 256), dim3(256), 0, stream,
                       x, Wg, bg, counts, eks, wks);
    hipLaunchKernelGGL(moe_offsets, dim3(1), dim3(32), 0, stream, counts, offsets);
    hipLaunchKernelGGL(moe_scatter, dim3((BT + 255) / 256), dim3(256), 0, stream,
                       eks, offsets, cursor, bucket, slotOf);
    // GEMM1: grid (F/BLK_N, BT/BLK_M, E); inactive tiles early-exit on counts
    hipLaunchKernelGGL(moe_ffn_gemm1, dim3(Fc / BLK_N, BT / BLK_M, Ec), dim3(256), 0, stream,
                       xh, W1t, b1, counts, offsets, bucket, H);
    // GEMM2: grid (D/BLK_N, BT/BLK_M, E)
    hipLaunchKernelGGL(moe_ffn_gemm2, dim3(Dc / BLK_N, BT / BLK_M, Ec), dim3(256), 0, stream,
                       H, W2t, counts, offsets, Yb);
    hipLaunchKernelGGL(moe_combine, dim3((BT * (Dc / 4) + 255) / 256), dim3(256), 0, stream,
                       Yb, b2, eks, wks, slotOf, out);
}
